// MultiHeadRegionalAttention_16836271800932
// MI455X (gfx1250) — matrix operand order repaired and checked by reading, not compiled, not run on hardware
//
#include <hip/hip_runtime.h>

typedef __attribute__((ext_vector_type(8)))  float          v8f;
typedef __attribute__((ext_vector_type(16))) __bf16         v16bf;
typedef __attribute__((ext_vector_type(4)))  unsigned short ush4;
typedef __attribute__((ext_vector_type(8)))  unsigned short ush8;
typedef __attribute__((ext_vector_type(16))) unsigned short ush16;

#define DEV __device__ __forceinline__

static constexpr int B_ = 2, N_ = 2048, E_ = 1024, H_ = 16, D_ = 64;
static constexpr int ROWS_ = B_ * N_;        // 4096
static constexpr float SCALE_ = 0.125f;      // 1/sqrt(64)

// ---------- helpers ----------
DEV unsigned short f2bf(float f) {
  unsigned int u = __builtin_bit_cast(unsigned int, f);
  u += 0x7FFFu + ((u >> 16) & 1u);           // round-to-nearest-even
  return (unsigned short)(u >> 16);
}

DEV v8f vzero8() {
  v8f z;
#pragma unroll
  for (int i = 0; i < 8; ++i) z[i] = 0.0f;
  return z;
}

DEV v16bf frag_cat(ush8 lo, ush8 hi) {
  ush16 t;
#pragma unroll
  for (int i = 0; i < 8; ++i) { t[i] = lo[i]; t[i + 8] = hi[i]; }
  return __builtin_bit_cast(v16bf, t);
}

// A-matrix 16x32 bf16 fragment. rowptr = &row[k_window_start].
// lanes 0-15: K = {0..7, 16..23}; lanes 16-31: K = {8..15, 24..31}
DEV v16bf load_a(const unsigned short* rowptr, int lane) {
  const int kb = (lane & 16) ? 8 : 0;
  ush8 lo = *(const ush8*)(rowptr + kb);
  ush8 hi = *(const ush8*)(rowptr + 16 + kb);
  return frag_cat(lo, hi);
}

// B-matrix 32x16 bf16 fragment (contraction contiguous in memory).
// colptr = &col[k_window_start]; lanes 0-15: K = {0..7, 16..23}; lanes 16-31: K = {8..15, 24..31}
DEV v16bf load_b(const unsigned short* colptr, int lane) {
  const int kb = (lane & 16) ? 8 : 0;
  ush8 lo = *(const ush8*)(colptr + kb);
  ush8 hi = *(const ush8*)(colptr + 16 + kb);
  return frag_cat(lo, hi);
}

DEV v8f wmma_bf16(v16bf a, v16bf b, v8f c) {
  return __builtin_amdgcn_wmma_f32_16x16x32_bf16(false, a, false, b,
                                                 (short)0, c, false, false);
}

// ---------- fp32 -> bf16 conversion ----------
__global__ __launch_bounds__(256) void cvt_bf16_kernel(
    const float* __restrict__ src, unsigned short* __restrict__ dst, int n4) {
  int i = blockIdx.x * blockDim.x + threadIdx.x;
  if (i < n4) {
    float4 v = ((const float4*)src)[i];
    ush4 o;
    o[0] = f2bf(v.x); o[1] = f2bf(v.y); o[2] = f2bf(v.z); o[3] = f2bf(v.w);
    ((ush4*)dst)[i] = o;
  }
}

// ---------- QKV projection: y = x @ W^T + b, written head-major (bf16) ----------
// transpose_out == 0 : out[((b*H+h)*N + n)*D + d]   (Q, K)
// transpose_out == 1 : out[((b*H+h)*D + d)*N + n]   (V, pre-transposed for P@V)
__global__ __launch_bounds__(256) void proj_qkv_kernel(
    const unsigned short* __restrict__ X, const unsigned short* __restrict__ W,
    const float* __restrict__ bias, unsigned short* __restrict__ out,
    int transpose_out) {
  const int lane = threadIdx.x & 31;
  const int wid  = threadIdx.x >> 5;
  const int tile = blockIdx.x * 8 + wid;         // 4096 tiles = 256 row x 16 col
  const int rt = tile >> 4, ct = tile & 15;
  const int i0 = rt * 16, o0 = ct * 64;
  const int col = lane & 15, half = lane >> 4;

  v8f acc[4];
#pragma unroll
  for (int g = 0; g < 4; ++g) acc[g] = vzero8();

  const unsigned short* xrow = X + (size_t)(i0 + col) * E_;  // A row = lane&15
  const unsigned short* wc[4];
#pragma unroll
  for (int g = 0; g < 4; ++g) wc[g] = W + (size_t)(o0 + g * 16 + col) * E_;

  auto load_ab = [&](int kk, v16bf& a, v16bf (&bb)[4]) {
    a = load_a(xrow + kk, lane);
#pragma unroll
    for (int g = 0; g < 4; ++g) bb[g] = load_b(wc[g] + kk, lane);
  };
  auto do_wmma = [&](const v16bf& a, const v16bf (&bb)[4]) {
#pragma unroll
    for (int g = 0; g < 4; ++g) acc[g] = wmma_bf16(a, bb[g], acc[g]);
  };

  // ping-pong double buffers: no register rotation copies
  v16bf aA, aB, bA[4], bB[4];
  load_ab(0, aA, bA);
  for (int kk = 0; kk < E_ - 64; kk += 64) {
    load_ab(kk + 32, aB, bB);
    do_wmma(aA, bA);
    load_ab(kk + 64, aA, bA);
    do_wmma(aB, bB);
  }
  load_ab(E_ - 32, aB, bB);
  do_wmma(aA, bA);
  do_wmma(aB, bB);

#pragma unroll
  for (int g = 0; g < 4; ++g) {
    const int o = o0 + g * 16 + col;
    const float bo = bias[o];
    const int h = o >> 6, d = o & 63;
#pragma unroll
    for (int r = 0; r < 8; ++r) {
      const int row = i0 + r + 8 * half;         // row = b*N + n
      const int b = row >> 11, n = row & (N_ - 1);
      const int bh = b * H_ + h;
      const float y = acc[g][r] + bo;
      size_t idx = transpose_out ? ((size_t)bh * D_ + d) * N_ + n
                                 : ((size_t)bh * N_ + n) * D_ + d;
      out[idx] = f2bf(y);
    }
  }
}

// ---------- flash attention, one wave per 16-query tile ----------
__global__ __launch_bounds__(128) void attn_kernel(
    const unsigned short* __restrict__ Qh, const unsigned short* __restrict__ Kh,
    const unsigned short* __restrict__ Vt, const unsigned char* __restrict__ mask,
    const float* __restrict__ region_mask, const float* __restrict__ region_bias,
    unsigned short* __restrict__ Ob) {
  __shared__ unsigned short pbuf[4][16 * 32];    // per-wave P staging (bf16)
  const int lane = threadIdx.x & 31;
  const int wid  = threadIdx.x >> 5;
  const int gt = blockIdx.x * 4 + wid;           // 4096 wave-tiles
  const int qt = gt & (N_ / 16 - 1);             // 0..127
  const int bh = gt >> 7;                        // 0..31
  const int b = bh >> 4, h = bh & 15;
  const int q0 = qt * 16;
  const int col = lane & 15, half = lane >> 4;

  const unsigned short* Qp = Qh + ((size_t)bh * N_ + q0) * D_;
  const unsigned short* Kp = Kh + (size_t)bh * N_ * D_;
  const unsigned short* Vp = Vt + (size_t)bh * D_ * N_;
  const unsigned char*  Mp = mask + ((size_t)b * N_ + q0) * N_;
  unsigned short* pl = pbuf[wid];

  float rbias[8];
  const float rb = region_bias[h];
#pragma unroll
  for (int r = 0; r < 8; ++r)
    rbias[r] = rb * region_mask[(size_t)b * N_ + q0 + r + 8 * half];

  const v16bf qf0 = load_a(Qp + (size_t)col * D_, lane);       // d 0..31
  const v16bf qf1 = load_a(Qp + (size_t)col * D_ + 32, lane);  // d 32..63

  v8f o_acc[4];
#pragma unroll
  for (int g = 0; g < 4; ++g) o_acc[g] = vzero8();
  float mrow[8], lrow[8];
#pragma unroll
  for (int r = 0; r < 8; ++r) { mrow[r] = -3.0e38f; lrow[r] = 0.0f; }

  // K fragments [c-group(0/1)][d-half(0/1)] ; V fragments per 16-wide d group
  auto load_kv = [&](int k0, v16bf (&kfr)[4], v16bf (&vfr)[4]) {
#pragma unroll
    for (int j = 0; j < 2; ++j) {
      kfr[2 * j + 0] = load_b(Kp + (size_t)(k0 + 16 * j + col) * D_, lane);
      kfr[2 * j + 1] = load_b(Kp + (size_t)(k0 + 16 * j + col) * D_ + 32, lane);
    }
#pragma unroll
    for (int g = 0; g < 4; ++g)
      vfr[g] = load_b(Vp + (size_t)(g * 16 + col) * N_ + k0, lane);
  };

  auto process_chunk = [&](int k0, const v16bf (&kfr)[4], const v16bf (&vfr)[4]) {
    // hoist mask byte loads so they overlap the WMMAs below
    unsigned char mk0[8], mk1[8];
#pragma unroll
    for (int r = 0; r < 8; ++r) {
      const size_t moff = (size_t)(r + 8 * half) * N_;
      mk0[r] = Mp[moff + k0 + col];
      mk1[r] = Mp[moff + k0 + 16 + col];
    }

    // S = Q K^T : 16 queries x 32 keys, contraction D=64 in two k=32 steps
    v8f s0 = vzero8(), s1 = vzero8();
    s0 = wmma_bf16(qf0, kfr[0], s0);
    s0 = wmma_bf16(qf1, kfr[1], s0);
    s1 = wmma_bf16(qf0, kfr[2], s1);
    s1 = wmma_bf16(qf1, kfr[3], s1);

#pragma unroll
    for (int r = 0; r < 8; ++r) {
      float a0 = s0[r] * SCALE_ + rbias[r];
      float a1 = s1[r] * SCALE_ + rbias[r];
      if (mk0[r]) a0 = -1.0e30f;
      if (mk1[r]) a1 = -1.0e30f;
      // row max across the 16 lanes holding this row
      float t = fmaxf(a0, a1);
      t = fmaxf(t, __shfl_xor(t, 1));
      t = fmaxf(t, __shfl_xor(t, 2));
      t = fmaxf(t, __shfl_xor(t, 4));
      t = fmaxf(t, __shfl_xor(t, 8));
      const float mn    = fmaxf(mrow[r], t);
      const float alpha = __expf(mrow[r] - mn);
      const float p0 = __expf(a0 - mn);
      const float p1 = __expf(a1 - mn);
      float rs = p0 + p1;
      rs += __shfl_xor(rs, 1);
      rs += __shfl_xor(rs, 2);
      rs += __shfl_xor(rs, 4);
      rs += __shfl_xor(rs, 8);
      lrow[r] = lrow[r] * alpha + rs;
      mrow[r] = mn;
#pragma unroll
      for (int g = 0; g < 4; ++g) o_acc[g][r] *= alpha;
      const int prow = (r + 8 * half) * 32;      // C-layout -> LDS [row][key]
      pl[prow + col]      = f2bf(p0);
      pl[prow + 16 + col] = f2bf(p1);
    }
    asm volatile("s_wait_dscnt 0" ::: "memory"); // P fully in LDS (same-wave RAW)
    const v16bf pf = load_a(pl + col * 32, lane);  // re-read P in A layout

    // O += P V : contraction over 32 keys, V^T fragments are contiguous
#pragma unroll
    for (int g = 0; g < 4; ++g) o_acc[g] = wmma_bf16(pf, vfr[g], o_acc[g]);
  };

  // ping-pong over 32-key chunks (2048 keys = 64 chunks)
  v16bf kA[4], vA[4], kB[4], vB[4];
  load_kv(0, kA, vA);
  for (int k0 = 0; k0 < N_ - 64; k0 += 64) {
    load_kv(k0 + 32, kB, vB);
    process_chunk(k0, kA, vA);
    load_kv(k0 + 64, kA, vA);
    process_chunk(k0 + 32, kB, vB);
  }
  load_kv(N_ - 32, kB, vB);
  process_chunk(N_ - 64, kA, vA);
  process_chunk(N_ - 32, kB, vB);

  // normalize and write bf16 back in (B, N, E) layout for the final projection
#pragma unroll
  for (int r = 0; r < 8; ++r) {
    const float inv = 1.0f / lrow[r];
    const int n = q0 + r + 8 * half;
    unsigned short* op = Ob + ((size_t)(b * N_ + n)) * E_ + h * D_;
#pragma unroll
    for (int g = 0; g < 4; ++g) op[g * 16 + col] = f2bf(o_acc[g][r] * inv);
  }
}

// ---------- output projection: d_out = O @ Wo^T + bo (fp32 out) ----------
__global__ __launch_bounds__(256) void proj_out_kernel(
    const unsigned short* __restrict__ X, const unsigned short* __restrict__ W,
    const float* __restrict__ bias, float* __restrict__ out) {
  const int lane = threadIdx.x & 31;
  const int wid  = threadIdx.x >> 5;
  const int tile = blockIdx.x * 8 + wid;
  const int rt = tile >> 4, ct = tile & 15;
  const int i0 = rt * 16, o0 = ct * 64;
  const int col = lane & 15, half = lane >> 4;

  v8f acc[4];
#pragma unroll
  for (int g = 0; g < 4; ++g) acc[g] = vzero8();

  const unsigned short* xrow = X + (size_t)(i0 + col) * E_;
  const unsigned short* wc[4];
#pragma unroll
  for (int g = 0; g < 4; ++g) wc[g] = W + (size_t)(o0 + g * 16 + col) * E_;

  auto load_ab = [&](int kk, v16bf& a, v16bf (&bb)[4]) {
    a = load_a(xrow + kk, lane);
#pragma unroll
    for (int g = 0; g < 4; ++g) bb[g] = load_b(wc[g] + kk, lane);
  };
  auto do_wmma = [&](const v16bf& a, const v16bf (&bb)[4]) {
#pragma unroll
    for (int g = 0; g < 4; ++g) acc[g] = wmma_bf16(a, bb[g], acc[g]);
  };

  v16bf aA, aB, bA[4], bB[4];
  load_ab(0, aA, bA);
  for (int kk = 0; kk < E_ - 64; kk += 64) {
    load_ab(kk + 32, aB, bB);
    do_wmma(aA, bA);
    load_ab(kk + 64, aA, bA);
    do_wmma(aB, bB);
  }
  load_ab(E_ - 32, aB, bB);
  do_wmma(aA, bA);
  do_wmma(aB, bB);

#pragma unroll
  for (int g = 0; g < 4; ++g) {
    const int o = o0 + g * 16 + col;
    const float bo = bias[o];
#pragma unroll
    for (int r = 0; r < 8; ++r) {
      const int row = i0 + r + 8 * half;
      out[(size_t)row * E_ + o] = acc[g][r] + bo;
    }
  }
}

// ---------- host launcher ----------
extern "C" void kernel_launch(void* const* d_in, const int* in_sizes, int n_in,
                              void* d_out, int out_size, void* d_ws, size_t ws_size,
                              hipStream_t stream) {
  const float* x           = (const float*)d_in[0];
  const float* region_mask = (const float*)d_in[1];
  const unsigned char* am  = (const unsigned char*)d_in[2];  // bool mask
  const float* Wq = (const float*)d_in[3];
  const float* bq = (const float*)d_in[4];
  const float* Wk = (const float*)d_in[5];
  const float* bk = (const float*)d_in[6];
  const float* Wv = (const float*)d_in[7];
  const float* bv = (const float*)d_in[8];
  const float* Wo = (const float*)d_in[9];
  const float* bo = (const float*)d_in[10];
  const float* region_bias = (const float*)d_in[11];

  char* ws = (char*)d_ws;
  const size_t XB_ELEMS = (size_t)ROWS_ * E_;    // 4 Mi elems
  const size_t W_ELEMS  = (size_t)E_ * E_;       // 1 Mi elems
  unsigned short* xb  = (unsigned short*)(ws);
  unsigned short* wqb = (unsigned short*)(ws + 8u  * 1024 * 1024);
  unsigned short* wkb = (unsigned short*)(ws + 10u * 1024 * 1024);
  unsigned short* wvb = (unsigned short*)(ws + 12u * 1024 * 1024);
  unsigned short* wob = (unsigned short*)(ws + 14u * 1024 * 1024);
  unsigned short* qh  = (unsigned short*)(ws + 16u * 1024 * 1024);
  unsigned short* kh  = (unsigned short*)(ws + 24u * 1024 * 1024);
  unsigned short* vth = (unsigned short*)(ws + 32u * 1024 * 1024);
  unsigned short* ob  = (unsigned short*)(ws + 40u * 1024 * 1024);

  // fp32 -> bf16
  cvt_bf16_kernel<<<(int)(XB_ELEMS / 4 / 256), 256, 0, stream>>>(x,  xb,  (int)(XB_ELEMS / 4));
  cvt_bf16_kernel<<<(int)(W_ELEMS  / 4 / 256), 256, 0, stream>>>(Wq, wqb, (int)(W_ELEMS / 4));
  cvt_bf16_kernel<<<(int)(W_ELEMS  / 4 / 256), 256, 0, stream>>>(Wk, wkb, (int)(W_ELEMS / 4));
  cvt_bf16_kernel<<<(int)(W_ELEMS  / 4 / 256), 256, 0, stream>>>(Wv, wvb, (int)(W_ELEMS / 4));
  cvt_bf16_kernel<<<(int)(W_ELEMS  / 4 / 256), 256, 0, stream>>>(Wo, wob, (int)(W_ELEMS / 4));

  // projections: 4096 wave-tiles, 8 waves/block
  proj_qkv_kernel<<<512, 256, 0, stream>>>(xb, wqb, bq, qh, 0);
  proj_qkv_kernel<<<512, 256, 0, stream>>>(xb, wkb, bk, kh, 0);
  proj_qkv_kernel<<<512, 256, 0, stream>>>(xb, wvb, bv, vth, 1);

  // flash attention: 4096 wave-tiles, 4 waves/block
  attn_kernel<<<1024, 128, 0, stream>>>(qh, kh, vth, am, region_mask,
                                        region_bias, ob);

  // output projection
  proj_out_kernel<<<512, 256, 0, stream>>>(ob, wob, bo, (float*)d_out);
}
